// FullyConnectedTensorProductConv_38886633897998
// MI455X (gfx1250) — matrix operand order repaired and checked by reading, not compiled, not run on hardware
//
#include <hip/hip_runtime.h>

// ---------------------------------------------------------------------------
// FullyConnectedTensorProductConv for MI455X (gfx1250, wave32, WMMA).
// Per-edge MLP (the 40-GFLOP hot spot) runs on v_wmma_f32_16x16x32_bf16 with
// 16 edges as the M dimension; the CG tensor product is fused into the W2
// GEMM epilogue so the 2304-wide per-edge weight tensor never hits memory.
// Hot loop: invariant A-fragments hoisted to registers, B tiles double-
// buffered so global fetch of tile t+1 overlaps the WMMA chain of tile t.
// ---------------------------------------------------------------------------

typedef __bf16 bf16_t;
typedef bf16_t v16bf __attribute__((ext_vector_type(16)));
typedef bf16_t v8bf  __attribute__((ext_vector_type(8)));
typedef float  v8f   __attribute__((ext_vector_type(8)));

#define M0_DIM   32
#define M1_DIM   16
#define IN_DIM   80          // 32 scalars + 16*3 vector components
#define HID      128
#define WN       2304        // tensor-product weight count per edge
#define TILE_E   16          // edges per workgroup (WMMA M)
#define THREADS  256         // 8 wave32 waves

__device__ __forceinline__ unsigned short f32_to_bf16_rne(float f) {
  union { float f; unsigned u; } v; v.f = f;
  unsigned r = v.u + 0x7FFFu + ((v.u >> 16) & 1u);
  return (unsigned short)(r >> 16);
}

__device__ __forceinline__ float gelu_exact(float x) {
  return 0.5f * x * (1.0f + erff(x * 0.70710678118654752f));
}

// A-fragment layout (matches B): lane holds row m = lane&15,
//   16 contiguous K starting at k0 + 16*(lane>>4).
// B-fragment: lane holds column n = lane&15, 16 contiguous K starting at
//   k0 + 16*(lane>>4)  -> one contiguous 32B load from row-major W^T rows.
struct AFrag { v16bf a[4]; };   // full K=128 of one 16x128 activation tile

__device__ __forceinline__ AFrag load_afrag(const unsigned short* __restrict__ hsrc,
                                            int mrow, int g) {
  AFrag f;
#pragma unroll
  for (int kk = 0; kk < 4; ++kk) {
    const int k0 = kk * 32;
    v8bf alo = *(const v8bf*)(hsrc + mrow * HID + k0 + 16 * g);
    v8bf ahi = *(const v8bf*)(hsrc + mrow * HID + k0 + 16 * g + 8);
    f.a[kk] = __builtin_shufflevector(alo, ahi, 0, 1, 2, 3, 4, 5, 6, 7,
                                      8, 9, 10, 11, 12, 13, 14, 15);
  }
  return f;
}

__device__ __forceinline__ void load_bfrag(v16bf b[4],
                                           const unsigned short* __restrict__ wrow,
                                           int mrow, int g) {
#pragma unroll
  for (int kk = 0; kk < 4; ++kk)
    b[kk] = *(const v16bf*)(wrow + mrow * HID + kk * 32 + 16 * g);
}

__device__ __forceinline__ v8f wmma_k128(const AFrag& f, const v16bf b[4], v8f acc) {
#pragma unroll
  for (int kk = 0; kk < 4; ++kk)
    acc = __builtin_amdgcn_wmma_f32_16x16x32_bf16(false, f.a[kk], false, b[kk],
                                                  (short)0, acc, false, false);
  return acc;
}

// ---------------------------------------------------------------------------
__global__ void zero_kernel(float* __restrict__ p, int n) {
  int i = blockIdx.x * 256 + threadIdx.x;
  if (i < n) p[i] = 0.0f;
}

__global__ void convert_weights_kernel(const float* __restrict__ W1,
                                       const float* __restrict__ W2,
                                       unsigned short* __restrict__ W1bf,
                                       unsigned short* __restrict__ W2bf) {
  int i = blockIdx.x * 256 + threadIdx.x;
  const int n1 = HID * HID;
  const int n2 = WN * HID;
  if (i < n1)            W1bf[i]      = f32_to_bf16_rne(W1[i]);
  else if (i < n1 + n2)  W2bf[i - n1] = f32_to_bf16_rne(W2[i - n1]);
}

// src_proj[n] = src_scalars[n] @ W0[:,32:48].T ; dst_proj likewise with 48:64
__global__ void nodeproj_kernel(const float* __restrict__ ss,
                                const float* __restrict__ ds,
                                const float* __restrict__ W0,
                                float* __restrict__ srcp,
                                float* __restrict__ dstp,
                                int Nsrc, int Ndst) {
  int i = blockIdx.x * 256 + threadIdx.x;
  int totS = Nsrc * HID;
  int tot  = totS + Ndst * HID;
  if (i >= tot) return;
  if (i < totS) {
    int n = i >> 7, o = i & 127;
    const float* w = &W0[o * 64 + 32];
    float s = 0.0f;
#pragma unroll
    for (int k = 0; k < 16; ++k) s += ss[n * 16 + k] * w[k];
    srcp[i] = s;
  } else {
    int j = i - totS;
    int n = j >> 7, o = j & 127;
    const float* w = &W0[o * 64 + 48];
    float s = 0.0f;
#pragma unroll
    for (int k = 0; k < 16; ++k) s += ds[n * 16 + k] * w[k];
    dstp[j] = s;
  }
}

// ---------------------------------------------------------------------------
__global__ __launch_bounds__(THREADS)
void edge_kernel(const float* __restrict__ xfeat,   // (Nsrc, 80)
                 const float* __restrict__ esh,     // (E, 4)
                 const float* __restrict__ eemb,    // (E, 32)
                 const float* __restrict__ W0,      // (128, 64) row-major
                 const float* __restrict__ b0,      // (128)
                 const float* __restrict__ b1,      // (128)
                 const float* __restrict__ b2,      // (2304)
                 const int*   __restrict__ sidx_g,  // (E)
                 const int*   __restrict__ didx_g,  // (E)
                 const float* __restrict__ srcp,    // (Nsrc, 128)
                 const float* __restrict__ dstp,    // (Ndst, 128)
                 const unsigned short* __restrict__ W1bf,  // (128,128) bf16
                 const unsigned short* __restrict__ W2bf,  // (2304,128) bf16
                 float* __restrict__ sums,          // (Ndst, 80) atomically acc
                 float* __restrict__ cnt,           // (Ndst)
                 int E) {
  __shared__ float s_xin[TILE_E * IN_DIM];      // gathered src features
  __shared__ float s_emb[TILE_E * 32];
  __shared__ float s_sh [TILE_E * 4];
  __shared__ float s_A0 [TILE_E * 32];          // x0*y0
  __shared__ float s_XY1[TILE_E * 32 * 3];      // x0*y1[c]
  __shared__ float s_XZ [TILE_E * 16 * 3];      // x1[:,c]*y0
  __shared__ float s_D11[TILE_E * 16];          // dot(x1,y1)/sqrt(3)
  __shared__ float s_out[TILE_E * IN_DIM];      // TP outputs (pre-scale)
  __shared__ float s_b2 [WN];
  __shared__ __align__(32) unsigned short s_h1[TILE_E * HID];
  __shared__ __align__(32) unsigned short s_h2[TILE_E * HID];
  __shared__ int s_si[TILE_E];
  __shared__ int s_di[TILE_E];

  const int tid = threadIdx.x;
  const int e0  = blockIdx.x * TILE_E;

  // -------- stage 1: cooperative staging into LDS ------------------------
  for (int i = tid; i < TILE_E * IN_DIM; i += THREADS) {
    int m = i / IN_DIM;
    int e = e0 + m; if (e >= E) e = E - 1;
    s_xin[i] = xfeat[(long)sidx_g[e] * IN_DIM + (i % IN_DIM)];
    s_out[i] = 0.0f;
  }
  for (int i = tid; i < TILE_E * 32; i += THREADS) {
    int m = i >> 5;
    int e = e0 + m; if (e >= E) e = E - 1;
    s_emb[i] = eemb[(long)e * 32 + (i & 31)];
  }
  if (tid < TILE_E * 4) {
    int m = tid >> 2;
    int e = e0 + m; if (e >= E) e = E - 1;
    s_sh[tid] = esh[(long)e * 4 + (tid & 3)];
  }
  if (tid < TILE_E) {
    int e = e0 + tid; if (e >= E) e = E - 1;
    s_si[tid] = sidx_g[e];
    s_di[tid] = didx_g[e];
  }
  for (int i = tid; i < WN; i += THREADS) s_b2[i] = b2[i];
  __syncthreads();

  // -------- stage 2: layer-1 MLP (K=32, VALU) + CG coefficient tables ----
  for (int i = tid; i < TILE_E * HID; i += THREADS) {
    int m = i >> 7, o = i & 127;
    float s = b0[o];
    const float* er = &s_emb[m * 32];
    const float* wr = &W0[o * 64];        // w_edge = W0[:, :32]
#pragma unroll
    for (int k = 0; k < 32; ++k) s += er[k] * wr[k];
    s += srcp[(long)s_si[m] * HID + o] + dstp[(long)s_di[m] * HID + o];
    s_h1[i] = f32_to_bf16_rne(gelu_exact(s));
  }
  for (int i = tid; i < TILE_E * 32; i += THREADS) {
    int m = i >> 5, k = i & 31;
    float x  = s_xin[m * IN_DIM + k];
    float y0 = s_sh[m * 4];
    s_A0[i]          = x * y0;
    s_XY1[i * 3 + 0] = x * s_sh[m * 4 + 1];
    s_XY1[i * 3 + 1] = x * s_sh[m * 4 + 2];
    s_XY1[i * 3 + 2] = x * s_sh[m * 4 + 3];
  }
  for (int i = tid; i < TILE_E * 16; i += THREADS) {
    int m = i >> 4, k = i & 15;
    float y0 = s_sh[m * 4];
    float v0 = s_xin[m * IN_DIM + 32 + 3 * k + 0];
    float v1 = s_xin[m * IN_DIM + 32 + 3 * k + 1];
    float v2 = s_xin[m * IN_DIM + 32 + 3 * k + 2];
    s_XZ[i * 3 + 0] = v0 * y0;
    s_XZ[i * 3 + 1] = v1 * y0;
    s_XZ[i * 3 + 2] = v2 * y0;
    s_D11[i] = 0.57735026918962576f *
               (v0 * s_sh[m * 4 + 1] + v1 * s_sh[m * 4 + 2] + v2 * s_sh[m * 4 + 3]);
  }
  __syncthreads();

  const int wv   = tid >> 5;     // wave id 0..7
  const int lane = tid & 31;
  const int mrow = lane & 15;    // WMMA column / A row
  const int g    = lane >> 4;    // lane half

  // -------- stage 3: h2 = gelu(h1 @ W1.T + b1) via WMMA ------------------
  {
    int n0 = wv * 16;
    AFrag af1 = load_afrag(s_h1, mrow, g);
    v16bf bw[4];
    load_bfrag(bw, W1bf + n0 * HID, mrow, g);
    v8f acc = {0.f, 0.f, 0.f, 0.f, 0.f, 0.f, 0.f, 0.f};
    acc = wmma_k128(af1, bw, acc);
#pragma unroll
    for (int v = 0; v < 8; ++v) {
      int m = v + 8 * g;                 // D layout: lane half selects row+8
      int n = n0 + mrow;
      s_h2[m * HID + n] = f32_to_bf16_rne(gelu_exact(acc[v] + b1[n]));
    }
  }
  __syncthreads();

  // -------- stage 4: tpw tiles (h2 @ W2.T + b2) fused with CG TP ---------
  // tpw channel blocks: [0,1024)=w00(32x32) [1024,1536)=w01(32x16)
  //                     [1536,1792)=w10(16x16) [1792,2304)=w11(16x32)
  float accLo[8], accHi[8], ac0[8], ac1[8], ac2[8];
#pragma unroll
  for (int v = 0; v < 8; ++v) { accLo[v] = accHi[v] = ac0[v] = ac1[v] = ac2[v] = 0.f; }

  // CG epilogue: consume one 16x16 tpw tile straight out of the accumulator.
  auto consume = [&](int t, const v8f& acc) {
    float bias = s_b2[t * 16 + mrow];
    if (t < 64) {                       // w00: out0[o] += (x0*y0)[i]*w
      int i = t >> 1;
      if (t & 1) {
#pragma unroll
        for (int v = 0; v < 8; ++v) {
          int m = v + 8 * g;
          accHi[v] += s_A0[m * 32 + i] * (acc[v] + bias);
        }
      } else {
#pragma unroll
        for (int v = 0; v < 8; ++v) {
          int m = v + 8 * g;
          accLo[v] += s_A0[m * 32 + i] * (acc[v] + bias);
        }
      }
    } else if (t < 96) {                // w01: out1[o][c] += x0[i]*y1[c]*w
      int i = t - 64;
#pragma unroll
      for (int v = 0; v < 8; ++v) {
        int m = v + 8 * g;
        float d = acc[v] + bias;
        const float* p = &s_XY1[(m * 32 + i) * 3];
        ac0[v] += p[0] * d; ac1[v] += p[1] * d; ac2[v] += p[2] * d;
      }
    } else if (t < 112) {               // w10: out1[o][c] += x1[i][c]*y0*w
      int i = t - 96;
#pragma unroll
      for (int v = 0; v < 8; ++v) {
        int m = v + 8 * g;
        float d = acc[v] + bias;
        const float* p = &s_XZ[(m * 16 + i) * 3];
        ac0[v] += p[0] * d; ac1[v] += p[1] * d; ac2[v] += p[2] * d;
      }
    } else {                            // w11: out0[o] += dot11[i]*w
      int tt = t - 112;
      int i = tt >> 1;
      if (tt & 1) {
#pragma unroll
        for (int v = 0; v < 8; ++v) {
          int m = v + 8 * g;
          accHi[v] += s_D11[m * 16 + i] * (acc[v] + bias);
        }
      } else {
#pragma unroll
        for (int v = 0; v < 8; ++v) {
          int m = v + 8 * g;
          accLo[v] += s_D11[m * 16 + i] * (acc[v] + bias);
        }
      }
    }
  };

  // 144 tiles, 18 per wave (uniform trip count -> EXEC stays all-ones).
  // A-fragments are loop-invariant: load once. B tiles are double-buffered
  // so the global fetch of tile t+8 overlaps the WMMA chain of tile t.
  {
    AFrag af2 = load_afrag(s_h2, mrow, g);
    v16bf bufA[4], bufB[4];
    load_bfrag(bufA, W2bf + (size_t)wv * 16 * HID, mrow, g);

    for (int it = 0; it < 18; it += 2) {
      const int t0 = wv + it * 8;
      const int t1 = t0 + 8;

      load_bfrag(bufB, W2bf + (size_t)t1 * 16 * HID, mrow, g);   // prefetch t1
      v8f acc0 = {0.f, 0.f, 0.f, 0.f, 0.f, 0.f, 0.f, 0.f};
      acc0 = wmma_k128(af2, bufA, acc0);
      consume(t0, acc0);

      if (it + 2 < 18)                                           // prefetch t2
        load_bfrag(bufA, W2bf + (size_t)(t0 + 16) * 16 * HID, mrow, g);
      v8f acc1 = {0.f, 0.f, 0.f, 0.f, 0.f, 0.f, 0.f, 0.f};
      acc1 = wmma_k128(af2, bufB, acc1);
      consume(t1, acc1);
    }
  }

  // cross-wave reduction in LDS (ds_add_f32)
#pragma unroll
  for (int v = 0; v < 8; ++v) {
    int m = v + 8 * g;
    atomicAdd(&s_out[m * IN_DIM + mrow],              accLo[v]);   // out0[0..15]
    atomicAdd(&s_out[m * IN_DIM + 16 + mrow],         accHi[v]);   // out0[16..31]
    atomicAdd(&s_out[m * IN_DIM + 32 + mrow * 3 + 0], ac0[v]);     // out1[o][0]
    atomicAdd(&s_out[m * IN_DIM + 32 + mrow * 3 + 1], ac1[v]);
    atomicAdd(&s_out[m * IN_DIM + 32 + mrow * 3 + 2], ac2[v]);
  }
  __syncthreads();

  // -------- stage 5: scaled scatter-add to destination nodes -------------
  const float scale = 0.14433756729740644f;   // 1/sqrt(48)
  for (int i = tid; i < TILE_E * IN_DIM; i += THREADS) {
    int m = i / IN_DIM;
    if (e0 + m < E)
      atomicAdd(&sums[(long)s_di[m] * IN_DIM + (i % IN_DIM)], s_out[i] * scale);
  }
  if (tid < TILE_E && (e0 + tid) < E)
    atomicAdd(&cnt[s_di[tid]], 1.0f);
}

// ---------------------------------------------------------------------------
__global__ void finalize_kernel(const float* __restrict__ sums,
                                const float* __restrict__ cnt,
                                float* __restrict__ out, int n) {
  int i = blockIdx.x * 256 + threadIdx.x;
  if (i < n) out[i] = sums[i] / fmaxf(cnt[i / IN_DIM], 1.0f);
}

// ---------------------------------------------------------------------------
extern "C" void kernel_launch(void* const* d_in, const int* in_sizes, int n_in,
                              void* d_out, int out_size, void* d_ws, size_t ws_size,
                              hipStream_t stream) {
  const float* xfeat = (const float*)d_in[0];   // src_features (Nsrc,80)
  const float* esh   = (const float*)d_in[1];   // edge_sh (E,4)
  const float* eemb  = (const float*)d_in[2];   // edge_emb (E,32)
  const float* ss    = (const float*)d_in[3];   // src_scalars (Nsrc,16)
  const float* ds    = (const float*)d_in[4];   // dst_scalars (Ndst,16)
  const float* W0    = (const float*)d_in[5];   // (128,64)
  const float* b0    = (const float*)d_in[6];
  const float* W1    = (const float*)d_in[7];   // (128,128)
  const float* b1    = (const float*)d_in[8];
  const float* W2    = (const float*)d_in[9];   // (2304,128)
  const float* b2    = (const float*)d_in[10];
  const int* sidx    = (const int*)d_in[11];
  const int* didx    = (const int*)d_in[12];

  const int Nsrc = in_sizes[0] / IN_DIM;
  const int Ndst = in_sizes[4] / 16;
  const int E    = in_sizes[11];

  // workspace carve-out (256B aligned)
  char* ws = (char*)d_ws;
  size_t off = 0;
  auto carve = [&](size_t bytes) -> void* {
    void* p = ws + off;
    off = (off + bytes + 255) & ~(size_t)255;
    return p;
  };
  float* sums = (float*)carve((size_t)Ndst * IN_DIM * sizeof(float));
  float* cnt  = (float*)carve((size_t)Ndst * sizeof(float));
  float* srcp = (float*)carve((size_t)Nsrc * HID * sizeof(float));
  float* dstp = (float*)carve((size_t)Ndst * HID * sizeof(float));
  unsigned short* W1bf = (unsigned short*)carve((size_t)HID * HID * 2);
  unsigned short* W2bf = (unsigned short*)carve((size_t)WN * HID * 2);

  // 1. zero accumulators
  {
    int n = Ndst * IN_DIM;
    zero_kernel<<<(n + 255) / 256, 256, 0, stream>>>(sums, n);
    zero_kernel<<<(Ndst + 255) / 256, 256, 0, stream>>>(cnt, Ndst);
  }
  // 2. bf16 weight conversion (row-major, K contiguous -> clean B fragments)
  {
    int n = HID * HID + WN * HID;
    convert_weights_kernel<<<(n + 255) / 256, 256, 0, stream>>>(W1, W2, W1bf, W2bf);
  }
  // 3. per-node scalar projections (reused ~4x per node)
  {
    int n = (Nsrc + Ndst) * HID;
    nodeproj_kernel<<<(n + 255) / 256, 256, 0, stream>>>(ss, ds, W0, srcp, dstp,
                                                         Nsrc, Ndst);
  }
  // 4. fused MLP + tensor product + scatter, 16 edges per workgroup
  {
    int blocks = (E + TILE_E - 1) / TILE_E;
    edge_kernel<<<blocks, THREADS, 0, stream>>>(xfeat, esh, eemb, W0, b0, b1, b2,
                                                sidx, didx, srcp, dstp, W1bf, W2bf,
                                                sums, cnt, E);
  }
  // 5. mean
  {
    int n = Ndst * IN_DIM;
    finalize_kernel<<<(n + 255) / 256, 256, 0, stream>>>(sums, cnt, (float*)d_out, n);
  }
}
